// ResBlock_6408091206411
// MI455X (gfx1250) — matrix operand order repaired and checked by reading, not compiled, not run on hardware
//
#include <hip/hip_runtime.h>
#include <hip/hip_bf16.h>

typedef __bf16 bf16_t;
typedef __attribute__((ext_vector_type(16))) __bf16 v16bf;
typedef __attribute__((ext_vector_type(8)))  float  v8f;

#define LRELU_SLOPE 0.01f
#define BN_EPS 1e-5f

#if defined(__HIP_DEVICE_COMPILE__) && defined(__gfx1250__) && \
    __has_builtin(__builtin_amdgcn_global_load_async_to_lds_b128) && \
    __has_builtin(__builtin_amdgcn_s_wait_asynccnt)
#define USE_ASYNC_LDS 1
#else
#define USE_ASYNC_LDS 0
#endif

#if USE_ASYNC_LDS
typedef __attribute__((vector_size(16))) int v4i_t;
typedef __attribute__((address_space(1))) v4i_t* gv4i_p;   // global src
typedef __attribute__((address_space(3))) v4i_t* lv4i_p;   // LDS dst
#endif

// ---------------------------------------------------------------------------
// Gather-GEMM sparse conv:  y[Nout,128] = sum_tap gather(src,kmap) @ Wt
//   Wt: pre-transposed bf16 [KTAPS][128(co)][CI]
//   block = 256 threads (8 waves), 32 output rows; wave w owns co tile w.
//   Each wave computes two 16x16 WMMA tiles (rows 0-15, 16-31) sharing B.
//   Double-buffered LDS A-tiles; gather for tap t+1 overlaps WMMAs of tap t.
//   Empty taps (idx<0) read a zeroed row (zrow) -> branch-free, EXEC stays 1s.
//   Async path: GLOBAL_LOAD_ASYNC_TO_LDS_B128 (ASYNCcnt), no VGPR round trip.
// do_act: LReLU + per-block BN partials (deterministic: channel owned by wave).
// ---------------------------------------------------------------------------
template <int CI, int KTAPS>
__global__ __launch_bounds__(256) void spconv_wmma(
    const bf16_t* __restrict__ src, const int* __restrict__ kmap,
    const bf16_t* __restrict__ Wt, const bf16_t* __restrict__ zrow,
    float* __restrict__ y, float* __restrict__ partial, int Nout, int do_act)
{
    constexpr int ROWS = 32;
    constexpr int CPR  = CI / 8;          // float4 chunks per gathered row
    constexpr int NCH  = ROWS * CPR;      // 256 (CI=64) or 512 (CI=128)
    constexpr int NPT  = NCH / 256;       // chunks per thread
    constexpr int NCIB = CI / 32;         // WMMA K-steps per tap
    constexpr int STR  = 136;             // LDS row stride (conflict-free pad)
    constexpr int RSH  = (CI == 64) ? 3 : 4;

    __shared__ __align__(16) bf16_t Atile[2][ROWS * STR];
    __shared__ float red[256];            // [0:128)=sum  [128:256)=sumsq
    __shared__ int   idxs[KTAPS * ROWS];

    const int tid  = threadIdx.x;
    const int lane = tid & 31;
    const int wave = tid >> 5;
    const int rr   = lane & 15;
    const int h    = lane >> 4;
    const int co   = wave * 16 + rr;
    const int r0   = blockIdx.x * ROWS;

    // Stage all tap indices for this 32-row tile
    for (int c = tid; c < KTAPS * ROWS; c += 256) {
        int tap = c >> 5, row = c & 31;
        int grow = r0 + row;
        idxs[c] = (grow < Nout) ? kmap[(size_t)tap * Nout + grow] : -1;
    }
    if (do_act) red[tid] = 0.0f;
    __syncthreads();

#if USE_ASYNC_LDS
    auto gather_async = [&](int tap, int buf) {
#pragma unroll
        for (int k = 0; k < NPT; ++k) {
            int c = tid + k * 256;
            int row = c >> RSH, part = c & (CPR - 1);
            int idx = idxs[tap * ROWS + row];
            const bf16_t* g = (idx >= 0) ? (src + (size_t)idx * CI + part * 8)
                                         : (zrow + part * 8);
            __builtin_amdgcn_global_load_async_to_lds_b128(
                (gv4i_p)(unsigned long long)g,
                (lv4i_p)(unsigned int)(unsigned long long)
                    (&Atile[buf][row * STR + part * 8]),
                0, 0);
        }
    };
#else
    float4 stage[NPT];
    auto gather = [&](int tap) {
#pragma unroll
        for (int k = 0; k < NPT; ++k) {
            int c = tid + k * 256;
            int row = c >> RSH, part = c & (CPR - 1);
            int idx = idxs[tap * ROWS + row];
            const bf16_t* g = (idx >= 0) ? (src + (size_t)idx * CI + part * 8)
                                         : (zrow + part * 8);
            stage[k] = *(const float4*)g;
        }
    };
    auto commit = [&](int buf) {
#pragma unroll
        for (int k = 0; k < NPT; ++k) {
            int c = tid + k * 256;
            int row = c >> RSH, part = c & (CPR - 1);
            *(float4*)(&Atile[buf][row * STR + part * 8]) = stage[k];
        }
    };
#endif

    v8f acc0 = {}, acc1 = {};

#if USE_ASYNC_LDS
    gather_async(0, 0);
    __builtin_amdgcn_s_wait_asynccnt(0);
    __syncthreads();
#else
    gather(0);
    commit(0);
    __syncthreads();
#endif

    for (int tap = 0; tap < KTAPS; ++tap) {
        const int cur = tap & 1;
#if USE_ASYNC_LDS
        if (tap + 1 < KTAPS) gather_async(tap + 1, cur ^ 1);  // DMA in flight
#else
        if (tap + 1 < KTAPS) gather(tap + 1);                 // loads in flight
#endif
        const bf16_t* wrow = Wt + ((size_t)tap * 128 + co) * CI + h * 8;
#pragma unroll
        for (int cib = 0; cib < NCIB; ++cib) {
            union { v16bf v; float4 f[2]; } A0, A1, B;
            const bf16_t* a0 = &Atile[cur][rr * STR + cib * 32 + h * 8];
            const bf16_t* a1 = a0 + 16 * STR;
            A0.f[0] = *(const float4*)(a0);          // K = cib*32 + h*8 ..
            A0.f[1] = *(const float4*)(a0 + 16);     // K = cib*32 + 16 + h*8 ..
            A1.f[0] = *(const float4*)(a1);
            A1.f[1] = *(const float4*)(a1 + 16);
            const bf16_t* bp = wrow + cib * 32;      // col=co, K = cib*32 + h*8 .. and cib*32 + 16 + h*8 ..
            B.f[0] = *(const float4*)(bp);
            B.f[1] = *(const float4*)(bp + 16);
#if defined(__HIP_DEVICE_COMPILE__)
            acc0 = __builtin_amdgcn_wmma_f32_16x16x32_bf16(
                false, A0.v, false, B.v, (short)0, acc0, false, false);
            acc1 = __builtin_amdgcn_wmma_f32_16x16x32_bf16(
                false, A1.v, false, B.v, (short)0, acc1, false, false);
#else
            (void)A0; (void)A1; (void)B;
#endif
        }
#if USE_ASYNC_LDS
        if (tap + 1 < KTAPS) __builtin_amdgcn_s_wait_asynccnt(0);
#else
        if (tap + 1 < KTAPS) commit(cur ^ 1);     // s_wait_loadcnt after WMMAs
#endif
        __syncthreads();
    }

    // Epilogue: LReLU, store both row tiles, BN partial stats
    float s = 0.f, q = 0.f;
#pragma unroll
    for (int i = 0; i < 8; ++i) {
        float v0 = acc0[i], v1 = acc1[i];
        if (do_act) {
            v0 = (v0 >= 0.f) ? v0 : LRELU_SLOPE * v0;
            v1 = (v1 >= 0.f) ? v1 : LRELU_SLOPE * v1;
        }
        int row0 = r0 + h * 8 + i;
        int row1 = row0 + 16;
        if (row0 < Nout) y[(size_t)row0 * 128 + co] = v0;
        if (row1 < Nout) y[(size_t)row1 * 128 + co] = v1;
        s += v0 + v1; q += v0 * v0 + v1 * v1;   // ghost rows contribute 0
    }
    if (do_act) {
        atomicAdd(&red[co], s);                 // each channel: one wave, 2 lanes
        atomicAdd(&red[128 + co], q);
        __syncthreads();
        partial[(size_t)blockIdx.x * 256 + tid] = red[tid];
    }
}

// ---------------------------------------------------------------------------
// Reduce per-block BN partials -> scale/bias per channel (grid = 128 blocks)
// ---------------------------------------------------------------------------
__global__ __launch_bounds__(256) void bn_finalize(
    const float* __restrict__ partial, int nblocks, float inv_count,
    const float* __restrict__ g, const float* __restrict__ b,
    float* __restrict__ sb)
{
    __shared__ float rs[256], rq[256];
    const int c = blockIdx.x;
    const int t = threadIdx.x;
    float s = 0.f, q = 0.f;
    for (int i = t; i < nblocks; i += 256) {
        s += partial[(size_t)i * 256 + c];
        q += partial[(size_t)i * 256 + 128 + c];
    }
    rs[t] = s; rq[t] = q;
    __syncthreads();
    for (int off = 128; off > 0; off >>= 1) {
        if (t < off) { rs[t] += rs[t + off]; rq[t] += rq[t + off]; }
        __syncthreads();
    }
    if (t == 0) {
        float mean  = rs[0] * inv_count;
        float var   = rq[0] * inv_count - mean * mean;
        float scale = g[c] * rsqrtf(var + BN_EPS);
        sb[c]       = scale;
        sb[128 + c] = b[c] - mean * scale;
    }
}

// ---------------------------------------------------------------------------
// Apply BN (y*scale+bias), optional residual add; x4 vectorized.
// outh gets 4 packed bf16 per thread (64-bit store).
// ---------------------------------------------------------------------------
__global__ __launch_bounds__(256) void bn_apply4(
    const float4* __restrict__ yv, const float* __restrict__ sb,
    const float4* __restrict__ addsrc, float4* __restrict__ outf,
    uint2* __restrict__ outh, size_t n4)
{
    size_t i = (size_t)blockIdx.x * 256 + threadIdx.x;
    if (i >= n4) return;
    int c = (int)((i * 4) & 127);
    float4 v = yv[i];
    v.x = v.x * sb[c + 0] + sb[128 + c + 0];
    v.y = v.y * sb[c + 1] + sb[128 + c + 1];
    v.z = v.z * sb[c + 2] + sb[128 + c + 2];
    v.w = v.w * sb[c + 3] + sb[128 + c + 3];
    if (addsrc) {
        float4 a = addsrc[i];
        v.x += a.x; v.y += a.y; v.z += a.z; v.w += a.w;
    }
    if (outf) outf[i] = v;
    if (outh) {
        union { bf16_t b[4]; uint2 u; } o;
        o.b[0] = (bf16_t)v.x; o.b[1] = (bf16_t)v.y;
        o.b[2] = (bf16_t)v.z; o.b[3] = (bf16_t)v.w;
        outh[i] = o.u;
    }
}

__global__ __launch_bounds__(256) void cvt_bf16x4(
    const float4* __restrict__ in, uint2* __restrict__ out, size_t n4)
{
    size_t i = (size_t)blockIdx.x * 256 + threadIdx.x;
    if (i >= n4) return;
    float4 v = in[i];
    union { bf16_t b[4]; uint2 u; } o;
    o.b[0] = (bf16_t)v.x; o.b[1] = (bf16_t)v.y;
    o.b[2] = (bf16_t)v.z; o.b[3] = (bf16_t)v.w;
    out[i] = o.u;
}

// W[K][Ci][Co] fp32 -> Wt[K][Co][Ci] bf16 (WMMA B-fragment friendly)
__global__ __launch_bounds__(256) void cvt_w(
    const float* __restrict__ W, bf16_t* __restrict__ Wt, int K, int Ci, int Co)
{
    int total = K * Ci * Co;
    int i = blockIdx.x * 256 + threadIdx.x;
    if (i >= total) return;
    int k  = i / (Ci * Co);
    int r  = i - k * Ci * Co;
    int ci = r / Co;
    int co = r - ci * Co;
    Wt[((size_t)k * Co + co) * Ci + ci] = (bf16_t)W[i];
}

__global__ void zero_u32(unsigned int* __restrict__ p, int n)
{
    int i = blockIdx.x * blockDim.x + threadIdx.x;
    if (i < n) p[i] = 0u;
}

// ---------------------------------------------------------------------------
extern "C" void kernel_launch(void* const* d_in, const int* in_sizes, int n_in,
                              void* d_out, int out_size, void* d_ws, size_t ws_size,
                              hipStream_t stream)
{
    const float* feats   = (const float*)d_in[0];
    const int*   map31   = (const int*)d_in[1];
    const int*   map13   = (const int*)d_in[2];
    const int*   mappool = (const int*)d_in[3];
    const float* W1      = (const float*)d_in[4];
    const float* W1_2    = (const float*)d_in[5];
    const float* W2      = (const float*)d_in[6];
    const float* W3      = (const float*)d_in[7];
    const float* Wpool   = (const float*)d_in[8];
    const float* g0   = (const float*)d_in[9];
    const float* b0   = (const float*)d_in[10];
    const float* g0_2 = (const float*)d_in[11];
    const float* b0_2 = (const float*)d_in[12];
    const float* g1   = (const float*)d_in[13];
    const float* b1   = (const float*)d_in[14];
    const float* g2   = (const float*)d_in[15];
    const float* b2   = (const float*)d_in[16];

    const int N = in_sizes[0] / 64;
    const int M = in_sizes[3] / 27;

    // scratch layout
    char* ws = (char*)d_ws;
    size_t off = 0;
    auto up = [](size_t x) { return (x + 255) & ~(size_t)255; };
    bf16_t* feats_h = (bf16_t*)(ws + off); off = up(off + (size_t)N * 64 * 2);
    bf16_t* Wt1  = (bf16_t*)(ws + off); off = up(off + (size_t)9 * 64 * 128 * 2);
    bf16_t* Wt12 = (bf16_t*)(ws + off); off = up(off + (size_t)9 * 128 * 128 * 2);
    bf16_t* Wt2  = (bf16_t*)(ws + off); off = up(off + (size_t)9 * 64 * 128 * 2);
    bf16_t* Wt3  = (bf16_t*)(ws + off); off = up(off + (size_t)9 * 128 * 128 * 2);
    bf16_t* Wtp  = (bf16_t*)(ws + off); off = up(off + (size_t)27 * 128 * 128 * 2);
    bf16_t* zrow = (bf16_t*)(ws + off); off = up(off + 128 * 2);
    float*  y    = (float*)(ws + off);  off = up(off + (size_t)N * 128 * 4);
    const int nbN = (N + 31) / 32;
    const int nbM = (M + 31) / 32;
    float*  partial = (float*)(ws + off); off = up(off + (size_t)nbN * 256 * 4);
    float*  sb   = (float*)(ws + off);  off = up(off + 256 * 4);
    bf16_t* buf16 = (bf16_t*)(ws + off); off = up(off + (size_t)N * 128 * 2);

    float* outB = (float*)d_out;                       // resB [M,128]
    float* outA = (float*)d_out + (size_t)M * 128;     // resA [N,128] (also shortcut)

    const size_t nF4 = (size_t)N * 64 / 4;
    const size_t nY  = (size_t)N * 128;
    const size_t nY4 = nY / 4;
    const float invN = 1.0f / (float)N;
    const int gY4 = (int)((nY4 + 255) / 256);

    // precision conversion + zero row
    zero_u32<<<1, 64, 0, stream>>>((unsigned int*)zrow, 64);
    cvt_bf16x4<<<(int)((nF4 + 255) / 256), 256, 0, stream>>>(
        (const float4*)feats, (uint2*)feats_h, nF4);
    cvt_w<<<(9 * 64 * 128 + 255) / 256, 256, 0, stream>>>(W1,   Wt1,  9, 64, 128);
    cvt_w<<<(9 * 128 * 128 + 255) / 256, 256, 0, stream>>>(W1_2, Wt12, 9, 128, 128);
    cvt_w<<<(9 * 64 * 128 + 255) / 256, 256, 0, stream>>>(W2,   Wt2,  9, 64, 128);
    cvt_w<<<(9 * 128 * 128 + 255) / 256, 256, 0, stream>>>(W3,   Wt3,  9, 128, 128);
    cvt_w<<<(27 * 128 * 128 + 255) / 256, 256, 0, stream>>>(Wpool, Wtp, 27, 128, 128);

    // conv1 (3,1,3): feats -> y ; BN -> buf16 (shortcut stage-1, bf16)
    spconv_wmma<64, 9><<<nbN, 256, 0, stream>>>(feats_h, map31, Wt1, zrow, y, partial, N, 1);
    bn_finalize<<<128, 256, 0, stream>>>(partial, nbN, invN, g0, b0, sb);
    bn_apply4<<<gY4, 256, 0, stream>>>((const float4*)y, sb, nullptr, nullptr,
                                       (uint2*)buf16, nY4);

    // conv1_2 (1,3,3): buf16 -> y ; BN -> shortcut fp32 in outA
    spconv_wmma<128, 9><<<nbN, 256, 0, stream>>>(buf16, map13, Wt12, zrow, y, partial, N, 1);
    bn_finalize<<<128, 256, 0, stream>>>(partial, nbN, invN, g0_2, b0_2, sb);
    bn_apply4<<<gY4, 256, 0, stream>>>((const float4*)y, sb, nullptr, (float4*)outA,
                                       nullptr, nY4);

    // conv2 (1,3,3): feats -> y ; BN -> buf16 (resA stage-1, bf16)
    spconv_wmma<64, 9><<<nbN, 256, 0, stream>>>(feats_h, map13, Wt2, zrow, y, partial, N, 1);
    bn_finalize<<<128, 256, 0, stream>>>(partial, nbN, invN, g1, b1, sb);
    bn_apply4<<<gY4, 256, 0, stream>>>((const float4*)y, sb, nullptr, nullptr,
                                       (uint2*)buf16, nY4);

    // conv3 (3,1,3): buf16 -> y ; BN + shortcut add -> resA (outA) + bf16 copy
    spconv_wmma<128, 9><<<nbN, 256, 0, stream>>>(buf16, map31, Wt3, zrow, y, partial, N, 1);
    bn_finalize<<<128, 256, 0, stream>>>(partial, nbN, invN, g2, b2, sb);
    bn_apply4<<<gY4, 256, 0, stream>>>((const float4*)y, sb, (const float4*)outA,
                                       (float4*)outA, (uint2*)buf16, nY4);

    // strided pool conv (3,3,3): buf16 -> resB (no activation / BN)
    spconv_wmma<128, 27><<<nbM, 256, 0, stream>>>(buf16, mappool, Wtp, zrow, outB, partial, M, 0);
}
